// CausalMultiheadSelfAttention_75385265979850
// MI455X (gfx1250) — matrix operand order repaired and checked by reading, not compiled, not run on hardware
//
#include <hip/hip_runtime.h>

typedef __bf16 bf16;
typedef __attribute__((ext_vector_type(16))) __bf16 v16bf;
typedef __attribute__((ext_vector_type(8)))  float  v8f;

#define D_MODEL 1024
#define T_SEQ   2048
#define BATCH   2
#define NH      16
#define DK      64
#define M_TOT   (BATCH * T_SEQ)   // 4096

union Frag16 { v16bf v; uint4 q[2]; };
union FragF  { v8f v; float f[8]; };

__device__ __forceinline__ bf16 f32_to_bf16(float x) {
  unsigned u = __builtin_bit_cast(unsigned, x);
  u += 0x7FFFu + ((u >> 16) & 1u);                  // round-to-nearest-even
  unsigned short h = (unsigned short)(u >> 16);
  return __builtin_bit_cast(bf16, h);
}

// A fragment: 16(M) x 32(K), 16-bit. row = lane%16.
// elems 0..7  <- K = (lane>=16 ? 8 : 0) + 0..7
// elems 8..15 <- K = 16 + (lane>=16 ? 8 : 0) + 0..7
__device__ __forceinline__ v16bf load_a(const bf16* p0, int ld, int lane) {
  const bf16* p = p0 + (size_t)(lane & 15) * ld + ((lane >> 4) << 3);
  Frag16 f;
  f.q[0] = *(const uint4*)(p);
  f.q[1] = *(const uint4*)(p + 16);
  return f.v;
}

// B fragment: 32(K) x 16(N), 16-bit, source stored as 16 rows (N) x K row-major.
// col = lane%16; elems 0..7 <- K = (lane>=16 ? 8 : 0) + 0..7, elems 8..15 <- K = 16 + (lane>=16 ? 8 : 0) + 0..7
__device__ __forceinline__ v16bf load_b(const bf16* p0, int ld, int lane) {
  const bf16* p = p0 + (size_t)(lane & 15) * ld + ((lane >> 4) << 3);
  Frag16 f;
  f.q[0] = *(const uint4*)(p);
  f.q[1] = *(const uint4*)(p + 16);
  return f.v;
}

__device__ __forceinline__ v8f wmma_bf16(v16bf a, v16bf b, v8f c) {
  return __builtin_amdgcn_wmma_f32_16x16x32_bf16(false, a, false, b,
                                                 (short)0, c, false, false);
}

// ---------------------------------------------------------------- converts
__global__ void cvt_f32_to_bf16(const float* __restrict__ src,
                                bf16* __restrict__ dst, int n) {
  int i = blockIdx.x * blockDim.x + threadIdx.x;
  int stride = gridDim.x * blockDim.x;
  for (; i < n; i += stride) dst[i] = f32_to_bf16(src[i]);
}

// ---------------------------------------------------------------- GEMM
// y[m,n] = sum_k A[m,k] * W[n,k]   (A: M_TOT x 1024 bf16, W: 1024 x 1024 bf16)
// Block 128 threads = 4 waves (2x2), wave tile 64x64 -> block tile 128x128.
// Per k-step: 8 fragment loads feed 16 WMMAs (1:1 VMEM:WMMA, 16 indep chains).
// mode 0: write bf16 to (b,h,t,d) layout      (Q,K projections)
// mode 1: write bf16 to (b,h,d,t) layout      (V projection, pre-transposed)
// mode 2: write f32 identity (m,n)            (output projection)
__global__ __launch_bounds__(128) void gemm_x_wT(const bf16* __restrict__ A,
                                                 const bf16* __restrict__ W,
                                                 void* __restrict__ Dst,
                                                 int mode) {
  const int lane = threadIdx.x & 31;
  const int wave = threadIdx.x >> 5;
  const int m0 = blockIdx.x * 128 + (wave >> 1) * 64;
  const int n0 = blockIdx.y * 128 + (wave & 1) * 64;
  FragF acc[4][4] = {};
  for (int k = 0; k < D_MODEL; k += 32) {
    v16bf a[4], b[4];
#pragma unroll
    for (int i = 0; i < 4; ++i)
      a[i] = load_a(A + (size_t)(m0 + i * 16) * D_MODEL + k, D_MODEL, lane);
#pragma unroll
    for (int i = 0; i < 4; ++i)
      b[i] = load_b(W + (size_t)(n0 + i * 16) * D_MODEL + k, D_MODEL, lane);
#pragma unroll
    for (int im = 0; im < 4; ++im)
#pragma unroll
      for (int in = 0; in < 4; ++in)
        acc[im][in].v = wmma_bf16(a[im], b[in], acc[im][in].v);
  }
  const int coln = lane & 15;
  const int half = lane >> 4;
#pragma unroll
  for (int im = 0; im < 4; ++im)
#pragma unroll
    for (int in = 0; in < 4; ++in)
#pragma unroll
      for (int j = 0; j < 8; ++j) {
        int m = m0 + im * 16 + half * 8 + j;
        int n = n0 + in * 16 + coln;
        float val = acc[im][in].f[j];
        if (mode == 2) {
          ((float*)Dst)[(size_t)m * D_MODEL + n] = val;
        } else {
          int bb = m >> 11, tt = m & (T_SEQ - 1);
          int hh = n >> 6, dd = n & (DK - 1);
          size_t idx = (mode == 0)
              ? ((((size_t)bb * NH + hh) * T_SEQ + tt) * DK + dd)
              : ((((size_t)bb * NH + hh) * DK + dd) * T_SEQ + tt);
          ((bf16*)Dst)[idx] = f32_to_bf16(val);
        }
      }
}

// ---------------------------------------------------------------- flash attention
// grid (T/64, B*NH), 128 threads. Wave w owns 16 query rows.
// Q,K: (b,h,t,d) bf16.  Vt: (b,h,d,t) bf16.  Y out: (b*t, 1024) bf16.
__global__ __launch_bounds__(128) void flash_attn(const bf16* __restrict__ Q,
                                                  const bf16* __restrict__ K,
                                                  const bf16* __restrict__ Vt,
                                                  bf16* __restrict__ Y) {
  __shared__ bf16 p_lds[4][16][80];   // per-wave P tile, 16B-aligned rows
  const int lane = threadIdx.x & 31;
  const int wave = threadIdx.x >> 5;
  const int coln = lane & 15;
  const int half = lane >> 4;
  const int bh = blockIdx.y;
  const int bb = bh >> 4, hh = bh & 15;
  const int qblk = blockIdx.x;
  const int q0 = qblk * 64 + wave * 16;

  const bf16* Qh = Q + (size_t)bh * T_SEQ * DK;
  const bf16* Kh = K + (size_t)bh * T_SEQ * DK;
  const bf16* Vh = Vt + (size_t)bh * DK * T_SEQ;

  v16bf qf[2];
  qf[0] = load_a(Qh + (size_t)q0 * DK + 0, DK, lane);
  qf[1] = load_a(Qh + (size_t)q0 * DK + 32, DK, lane);

  FragF o[4] = {};
  float mrow[8], lrow[8];
#pragma unroll
  for (int j = 0; j < 8; ++j) { mrow[j] = -3.0e38f; lrow[j] = 0.0f; }

  for (int kb = 0; kb <= qblk; ++kb) {
    // prefetch next key block's K tile (8KB) and V rows while this block runs
    if (kb < qblk) {
      const bf16* nk = Kh + (size_t)(kb + 1) * 64 * DK;
      const bf16* nv = Vh + (size_t)(kb + 1) * 64;
      if (wave == 0) __builtin_prefetch(nk + (size_t)lane * 128, 0, 0);
      else if (wave == 1) __builtin_prefetch(nv + (size_t)lane * T_SEQ, 0, 0);
      else if (wave == 2) __builtin_prefetch(nv + (size_t)(lane + 32) * T_SEQ, 0, 0);
    }
    // ---- S = (Q Kt) * 1/sqrt(dk)
    FragF s[4] = {};
#pragma unroll
    for (int ks = 0; ks < 2; ++ks)
#pragma unroll
      for (int nt = 0; nt < 4; ++nt) {
        v16bf kf = load_b(Kh + (size_t)(kb * 64 + nt * 16) * DK + ks * 32,
                          DK, lane);
        s[nt].v = wmma_bf16(qf[ks], kf, s[nt].v);
      }
    const bool diag = (kb == qblk);   // mask only possible on diagonal block
    float mnew[8];
#pragma unroll
    for (int j = 0; j < 8; ++j) {
      int r = q0 + half * 8 + j;
      float mx = -3.0e38f;
#pragma unroll
      for (int nt = 0; nt < 4; ++nt) {
        float sv = s[nt].f[j] * 0.125f;
        if (diag && (kb * 64 + nt * 16 + coln > r)) sv = -3.0e38f;
        s[nt].f[j] = sv;
        mx = fmaxf(mx, sv);
      }
      mx = fmaxf(mx, __shfl_xor(mx, 1, 32));
      mx = fmaxf(mx, __shfl_xor(mx, 2, 32));
      mx = fmaxf(mx, __shfl_xor(mx, 4, 32));
      mx = fmaxf(mx, __shfl_xor(mx, 8, 32));   // stays within 16-lane half
      mnew[j] = fmaxf(mx, mrow[j]);
    }
#pragma unroll
    for (int j = 0; j < 8; ++j) {
      float corr = __expf(mrow[j] - mnew[j]);
      float rs = 0.0f;
#pragma unroll
      for (int nt = 0; nt < 4; ++nt) {
        float p = __expf(s[nt].f[j] - mnew[j]);
        s[nt].f[j] = p;
        rs += p;
      }
      rs += __shfl_xor(rs, 1, 32);
      rs += __shfl_xor(rs, 2, 32);
      rs += __shfl_xor(rs, 4, 32);
      rs += __shfl_xor(rs, 8, 32);
      lrow[j] = lrow[j] * corr + rs;
      mrow[j] = mnew[j];
#pragma unroll
      for (int nt = 0; nt < 4; ++nt) o[nt].f[j] *= corr;
    }
    // ---- stage P (D-layout) -> LDS -> A-fragment layout (same-wave, DS in-order)
#pragma unroll
    for (int nt = 0; nt < 4; ++nt)
#pragma unroll
      for (int j = 0; j < 8; ++j)
        p_lds[wave][half * 8 + j][nt * 16 + coln] = f32_to_bf16(s[nt].f[j]);
    // ---- O += P V  (V pre-transposed: contiguous along key dim)
#pragma unroll
    for (int ks = 0; ks < 2; ++ks) {
      v16bf pf = load_a(&p_lds[wave][0][ks * 32], 80, lane);
#pragma unroll
      for (int nt = 0; nt < 4; ++nt) {
        v16bf vf = load_b(Vh + (size_t)(nt * 16) * T_SEQ + (size_t)kb * 64 +
                              ks * 32,
                          T_SEQ, lane);
        o[nt].v = wmma_bf16(pf, vf, o[nt].v);
      }
    }
  }
  // ---- normalize and write Y in (b*t, 1024) row-major (head-interleaved cols)
#pragma unroll
  for (int nt = 0; nt < 4; ++nt)
#pragma unroll
    for (int j = 0; j < 8; ++j) {
      int r = q0 + half * 8 + j;
      float val = o[nt].f[j] / lrow[j];
      Y[(size_t)(bb * T_SEQ + r) * D_MODEL + hh * DK + nt * 16 + coln] =
          f32_to_bf16(val);
    }
}

// ---------------------------------------------------------------- launch
extern "C" void kernel_launch(void* const* d_in, const int* in_sizes, int n_in,
                              void* d_out, int out_size, void* d_ws,
                              size_t ws_size, hipStream_t stream) {
  const float* x  = (const float*)d_in[0];
  const float* wq = (const float*)d_in[1];
  const float* wk = (const float*)d_in[2];
  const float* wv = (const float*)d_in[3];
  const float* wo = (const float*)d_in[4];
  float* out = (float*)d_out;

  char* ws = (char*)d_ws;
  size_t off = 0;
  auto carve = [&](size_t bytes) -> void* {
    void* p = ws + off;
    off += (bytes + 255) & ~(size_t)255;
    return p;
  };
  const size_t NX = (size_t)M_TOT * D_MODEL;     // 4M elements
  const size_t NW = (size_t)D_MODEL * D_MODEL;   // 1M elements
  bf16* xb  = (bf16*)carve(NX * 2);
  bf16* wqb = (bf16*)carve(NW * 2);
  bf16* wkb = (bf16*)carve(NW * 2);
  bf16* wvb = (bf16*)carve(NW * 2);
  bf16* wob = (bf16*)carve(NW * 2);
  bf16* Qb  = (bf16*)carve(NX * 2);              // (b,h,t,d)
  bf16* Kb  = (bf16*)carve(NX * 2);              // (b,h,t,d)
  bf16* Vtb = (bf16*)carve(NX * 2);              // (b,h,d,t)
  bf16* Yb  = (bf16*)carve(NX * 2);              // (b*t, d_model)

  cvt_f32_to_bf16<<<2048, 256, 0, stream>>>(x, xb, (int)NX);
  cvt_f32_to_bf16<<<1024, 256, 0, stream>>>(wq, wqb, (int)NW);
  cvt_f32_to_bf16<<<1024, 256, 0, stream>>>(wk, wkb, (int)NW);
  cvt_f32_to_bf16<<<1024, 256, 0, stream>>>(wv, wvb, (int)NW);
  cvt_f32_to_bf16<<<1024, 256, 0, stream>>>(wo, wob, (int)NW);

  dim3 gg(M_TOT / 128, D_MODEL / 128);           // 32 x 8
  gemm_x_wT<<<gg, 128, 0, stream>>>(xb, wqb, (void*)Qb, 0);
  gemm_x_wT<<<gg, 128, 0, stream>>>(xb, wkb, (void*)Kb, 0);
  gemm_x_wT<<<gg, 128, 0, stream>>>(xb, wvb, (void*)Vtb, 1);

  flash_attn<<<dim3(T_SEQ / 64, BATCH * NH), 128, 0, stream>>>(Qb, Kb, Vtb, Yb);

  gemm_x_wT<<<gg, 128, 0, stream>>>(Yb, wob, (void*)out, 2);
}
